// Block_8641474199860
// MI455X (gfx1250) — matrix operand order repaired and checked by reading, not compiled, not run on hardware
//
#include <hip/hip_runtime.h>
#include <hip/hip_bf16.h>
#include <math.h>

// Problem constants (match reference)
#define CB   2
#define CT   2048
#define CD   768
#define CH   12
#define CDH  64
#define CM   (CB*CT)      // 4096 rows for dense GEMMs

typedef __attribute__((ext_vector_type(16))) __bf16 bf16x16;
typedef __attribute__((ext_vector_type(8)))  __bf16 bf16x8;
typedef __attribute__((ext_vector_type(8)))  float  f32x8;
typedef __attribute__((ext_vector_type(4)))  int    i32x4;

// CDNA5 async global->LDS DMA (ASYNCcnt-tracked). Guarded so the file
// compiles on toolchains that lack the builtins (falls back to ds stores).
#if defined(__AMDGCN__) && \
    __has_builtin(__builtin_amdgcn_global_load_async_to_lds_b128) && \
    __has_builtin(__builtin_amdgcn_s_wait_asynccnt)
#define HAVE_ASYNC_LDS 1
typedef __attribute__((address_space(1))) i32x4 gi32x4;  // global int4
typedef __attribute__((address_space(3))) i32x4 li32x4;  // LDS int4
#else
#define HAVE_ASYNC_LDS 0
#endif

// CDNA5 Tensor Data Mover (TENSORcnt-tracked descriptor DMA).
#if defined(__AMDGCN__) && \
    __has_builtin(__builtin_amdgcn_tensor_load_to_lds) && \
    __has_builtin(__builtin_amdgcn_s_wait_tensorcnt)
#define HAVE_TDM 1
typedef __attribute__((ext_vector_type(4))) unsigned int u32x4;
typedef __attribute__((ext_vector_type(8))) int          i32x8;
#else
#define HAVE_TDM 0
#endif

static __device__ __forceinline__ void cp16_g2l(const __bf16* g, __bf16* l) {
#if HAVE_ASYNC_LDS
  __builtin_amdgcn_global_load_async_to_lds_b128((gi32x4*)g, (li32x4*)l, 0, 0);
#else
  *(uint4*)l = *(const uint4*)g;
#endif
}

static __device__ __forceinline__ void wait_async0() {
#if HAVE_ASYNC_LDS
  __builtin_amdgcn_s_wait_asynccnt(0);
#endif
}

#if HAVE_TDM
// Raw LDS byte offset of a pointer into __shared__ memory.
static __device__ __forceinline__ unsigned lds_offset(const void* p) {
  return (unsigned)(size_t)(__attribute__((address_space(3))) const void*)p;
}

// TDM 2D tile load: `rows` x 32 bf16 elements from row-major source with
// element row-stride K, into LDS with a 40-element row pitch (64B data +
// 16B pad per row via D# pad_enable). Issued by one wave; TENSORcnt-tracked.
// D# layout per CDNA5 ISA 8.3/8.4 (group0: ctrl/lds/global/type; group1:
// data_size, pad, tensor dims/strides, tile dims; groups 2-3 zero for 2D).
// This toolchain exposes the 6-argument builtin form:
//   (uint32x4 g0, int32x8 g1, int32x4, int32x4, int32x8, i32 cpol)
static __device__ __forceinline__ void tdm_load_tile(const __bf16* gsrc,
                                                     const __bf16* lds,
                                                     int rows, int K) {
  union { u32x4 v; unsigned long long q[2]; } g0;
  union { i32x8 v; unsigned long long q[4]; } g1;
  unsigned long long ga = (unsigned long long)(size_t)gsrc;
  g0.q[0] = 1ull                                            // count=1 (valid)
          | ((unsigned long long)lds_offset(lds) << 32);    // lds_addr
  g0.q[1] = (ga & 0x01FFFFFFFFFFFFFFull)                    // global_addr[56:0]
          | (2ull << 62);                                   // type=2 ("image")
  unsigned ctrl = (1u << 16)    // data_size: 2 bytes
                | (1u << 20)    // pad_enable
                | (3u << 22)    // pad_interval: 16 DWORDs (64B) between pads
                | (3u << 25);   // pad_amount: 4 DWORDs (16B)
  unsigned long long td0 = (unsigned)K;      // tensor_dim0 (elements)
  unsigned long long td1 = (unsigned)rows;   // tensor_dim1
  unsigned long long s0  = (unsigned)K;      // tensor_dim0_stride (elements)
  g1.q[0] = (unsigned long long)ctrl | ((td0 & 0xFFFFull) << 48);
  g1.q[1] = (td0 >> 16) | (td1 << 16) | (32ull << 48);      // tile_dim0 = 32
  g1.q[2] = (unsigned long long)(unsigned)rows              // tile_dim1 = rows
          | ((s0 & 0xFFFFFFFFull) << 32);
  g1.q[3] = (s0 >> 32) & 0xFFFFull;                         // dim1_stride = 0
  i32x4 z4 = {0, 0, 0, 0};
  i32x8 z8 = {0, 0, 0, 0, 0, 0, 0, 0};
  __builtin_amdgcn_tensor_load_to_lds(g0.v, g1.v, z4, z4, z8, 0);
}
#endif

// ---------------------------------------------------------------------------
// WMMA helpers (gfx1250, wave32). Fragment layouts per CDNA5 ISA 7.12.2:
//  A (16x32 bf16): lane l: row = l&15; VGPR0-3 hold K = (l>=16?8:0)+0..7,
//                  VGPR4-7 hold K = 16 + (l>=16?8:0)+0..7  -> two 16B chunks.
//  B (32x16 bf16): lane l: col = l&15; same K order as A:
//                  (l>=16?8:0)+0..7 and 16+(l>=16?8:0)+0..7 of B^T row (we store B^T).
//  C/D (16x16 f32): lane l: col = l&15; VGPR i: row = i + (l>=16?8:0).
// ---------------------------------------------------------------------------
static __device__ __forceinline__ bf16x16 wmma_ld_a(const __bf16* base, int stride) {
  const int l  = threadIdx.x & 31;
  const int r  = l & 15;
  const int hi = l >> 4;
  union { bf16x16 v; bf16x8 h[2]; } u;
  const __bf16* p = base + (size_t)r * stride + hi * 8;
  u.h[0] = *(const bf16x8*)(p);
  u.h[1] = *(const bf16x8*)(p + 16);
  return u.v;
}

static __device__ __forceinline__ bf16x16 wmma_ld_b(const __bf16* base, size_t stride) {
  const int l  = threadIdx.x & 31;
  const int c  = l & 15;
  const int hi = l >> 4;
  union { bf16x16 v; bf16x8 h[2]; } u;
  const __bf16* p = base + (size_t)c * stride + hi * 8;
  u.h[0] = *(const bf16x8*)(p);
  u.h[1] = *(const bf16x8*)(p + 16);
  return u.v;
}

static __device__ __forceinline__ f32x8 wmma_bf16(bf16x16 a, bf16x16 b, f32x8 c) {
  return __builtin_amdgcn_wmma_f32_16x16x32_bf16(false, a, false, b, (short)0, c,
                                                 false, false);
}

static __device__ __forceinline__ float gelu_f(float x) {
  const float c = 0.7978845608028654f;  // sqrt(2/pi)
  return 0.5f * x * (1.0f + tanhf(c * (x + 0.044715f * x * x * x)));
}

// ---------------------------------------------------------------------------
// Conversion kernels
// ---------------------------------------------------------------------------
__global__ __launch_bounds__(256) void cvt_bf16(const float* __restrict__ in,
                                                __bf16* __restrict__ out, int n) {
  int i = blockIdx.x * 256 + threadIdx.x;
  if (i < n) out[i] = (__bf16)in[i];
}

// in: [K][N] row-major f32  ->  out: [N][K] row-major bf16 (B^T for GEMM)
__global__ __launch_bounds__(256) void transpose_cvt(const float* __restrict__ in,
                                                     __bf16* __restrict__ out,
                                                     int K, int N) {
  int i = blockIdx.x * 256 + threadIdx.x;
  if (i < K * N) {
    int n = i / K, k = i % K;
    out[i] = (__bf16)in[(size_t)k * N + n];
  }
}

// ---------------------------------------------------------------------------
// WMMA GEMM: C[M,N] = A_bf16[M,K] @ (Bt_bf16[N,K])^T + bias
// Block tile 128x256, 8 waves (2x4), wave tile 64x64 (4x4 wmma), k-step 32.
// Double-buffered LDS. Per k-step: A tile streamed with per-lane async
// global->LDS DMA (ASYNCcnt); B (weight) tile moved by one TDM descriptor
// issued from wave 0 (TENSORcnt). One barrier per k-step.
// EPI: 0 = QKV split (q/k row-major per head, v transposed per head)
//      1/3 = write f32      2 = gelu -> bf16
// ---------------------------------------------------------------------------
template <int EPI>
__global__ __launch_bounds__(256, 1) void gemm256(
    const __bf16* __restrict__ A, const __bf16* __restrict__ Bt,
    const float* __restrict__ bias,
    float* __restrict__ outf, __bf16* __restrict__ outb,
    __bf16* __restrict__ qb, __bf16* __restrict__ kb, __bf16* __restrict__ vb,
    int M, int N, int K) {
  __shared__ __align__(16) __bf16 As[2][128 * 40];
  __shared__ __align__(16) __bf16 Bs[2][256 * 40];

  const int tid = threadIdx.x;
  const int bm = blockIdx.y, bn = blockIdx.x;
  const int wid = tid >> 5;
  const int wm = wid >> 2;   // 0..1  (64 rows each)
  const int wn = wid & 3;    // 0..3  (64 cols each)
  const int l = tid & 31, lc = l & 15, hi = l >> 4;

  f32x8 acc[4][4];
#pragma unroll
  for (int i = 0; i < 4; i++)
#pragma unroll
    for (int j = 0; j < 4; j++) acc[i][j] = {};

  // Stage one 32-deep k-slice of A(128 rows) and Bt(256 rows) into LDS buf.
  auto stage = [&](int buf, int k0) {
    // A tile: 512 x 16B per-lane async chunks
#pragma unroll
    for (int it = 0; it < 2; ++it) {
      int idx = tid + it * 256;
      int row = idx >> 2, ch = idx & 3;
      cp16_g2l(&A[(size_t)(bm * 128 + row) * K + k0 + ch * 8],
               &As[buf][row * 40 + ch * 8]);
    }
    // B tile: one TDM descriptor (wave 0) or per-lane fallback
#if HAVE_TDM
    if (wid == 0)
      tdm_load_tile(&Bt[(size_t)(bn * 256) * K + k0], &Bs[buf][0], 256, K);
#else
#pragma unroll
    for (int it = 0; it < 4; ++it) {
      int idx = tid + it * 256;
      int row = idx >> 2, ch = idx & 3;
      cp16_g2l(&Bt[(size_t)(bn * 256 + row) * K + k0 + ch * 8],
               &Bs[buf][row * 40 + ch * 8]);
    }
#endif
  };

  stage(0, 0);
  int cur = 0;
  for (int k0 = 0; k0 < K; k0 += 32) {
    wait_async0();        // this wave's A-tile DMA into buf[cur] done
#if HAVE_TDM
    if (wid == 0) __builtin_amdgcn_s_wait_tensorcnt(0);  // B-tile TDM done
#endif
    __syncthreads();      // everyone's data ready (and prev readers drained)
    if (k0 + 32 < K) stage(cur ^ 1, k0 + 32);

    bf16x16 af[4];
#pragma unroll
    for (int i = 0; i < 4; i++)
      af[i] = wmma_ld_a(&As[cur][(wm * 64 + i * 16) * 40], 40);
#pragma unroll
    for (int j = 0; j < 4; j++) {
      bf16x16 bb = wmma_ld_b(&Bs[cur][(wn * 64 + j * 16) * 40], 40);
#pragma unroll
      for (int i = 0; i < 4; i++) acc[i][j] = wmma_bf16(af[i], bb, acc[i][j]);
    }
    cur ^= 1;
  }

  // Epilogue
#pragma unroll
  for (int i = 0; i < 4; i++) {
#pragma unroll
    for (int j = 0; j < 4; j++) {
#pragma unroll
      for (int e = 0; e < 8; e++) {
        int row = bm * 128 + wm * 64 + i * 16 + hi * 8 + e;
        int col = bn * 256 + wn * 64 + j * 16 + lc;
        float v = acc[i][j][e] + bias[col];
        if (EPI == 0) {
          int b = row >> 11, t = row & (CT - 1);
          if (col < CD) {
            int h = col >> 6, d = col & 63;
            qb[(((size_t)b * CH + h) * CT + t) * CDH + d] = (__bf16)v;
          } else if (col < 2 * CD) {
            int cc = col - CD, h = cc >> 6, d = cc & 63;
            kb[(((size_t)b * CH + h) * CT + t) * CDH + d] = (__bf16)v;
          } else {
            int cc = col - 2 * CD, h = cc >> 6, d = cc & 63;
            // V stored transposed [B,H,DH,T] so P@V B-fragments are contiguous
            vb[(((size_t)b * CH + h) * CDH + d) * CT + t] = (__bf16)v;
          }
        } else if (EPI == 2) {
          outb[(size_t)row * N + col] = (__bf16)gelu_f(v);
        } else {
          outf[(size_t)row * N + col] = v;
        }
      }
    }
  }
}

// ---------------------------------------------------------------------------
// Flash attention: one wave per (b, h, 16-query tile). Online softmax over
// 32-key steps; probs relayout C-frag -> A-frag via per-wave LDS tile.
// Matches reference masking: invalid keys get -1e9 scores, invalid query
// rows are zeroed at the end (reference's softmax*amask semantics).
// ---------------------------------------------------------------------------
__global__ __launch_bounds__(256) void attn_kernel(
    const __bf16* __restrict__ qb, const __bf16* __restrict__ kb,
    const __bf16* __restrict__ vb, const int* __restrict__ lengths,
    __bf16* __restrict__ aout) {
  __shared__ __align__(16) __bf16 pt[8][16 * 40];

  const int tid = threadIdx.x;
  const int wid = tid >> 5;
  const int l = tid & 31, lc = l & 15, hi = l >> 4;

  int g  = blockIdx.x * 8 + wid;        // global wave index, 0..3071
  int qt = g & 127;                     // T/16 = 128 query tiles
  int h  = (g >> 7) % CH;
  int b  = g / (128 * CH);
  int m0 = qt * 16;
  int len = lengths[b];

  const __bf16* qbase = qb + (((size_t)b * CH + h) * CT + m0) * CDH;
  bf16x16 a0 = wmma_ld_a(qbase, CDH);        // dh 0..31
  bf16x16 a1 = wmma_ld_a(qbase + 32, CDH);   // dh 32..63

  f32x8 o[4];
#pragma unroll
  for (int j = 0; j < 4; j++) o[j] = {};
  float rm[8], rl[8];
#pragma unroll
  for (int i = 0; i < 8; i++) { rm[i] = -1e30f; rl[i] = 0.0f; }

  __bf16* myp = &pt[wid][0];

  const __bf16* kh = kb + ((size_t)b * CH + h) * CT * CDH;
  const __bf16* vh = vb + ((size_t)b * CH + h) * CDH * CT;

  for (int n0 = 0; n0 < CT; n0 += 32) {
    const __bf16* kbase = kh + (size_t)n0 * CDH;
    // Cover latency of the next tile (lands in WGP$/L2): global_prefetch_b8
    if (n0 + 32 < CT) {
      __builtin_prefetch(kh + (size_t)(n0 + 32) * CDH + lc * CDH, 0, 3);
      __builtin_prefetch(vh + (size_t)lc * CT + n0 + 32, 0, 3);
    }
    bf16x16 b00 = wmma_ld_b(kbase, CDH);
    bf16x16 b01 = wmma_ld_b(kbase + 32, CDH);
    bf16x16 b10 = wmma_ld_b(kbase + 16 * CDH, CDH);
    bf16x16 b11 = wmma_ld_b(kbase + 16 * CDH + 32, CDH);

    f32x8 s0 = {}, s1 = {};
    s0 = wmma_bf16(a0, b00, s0);
    s0 = wmma_bf16(a1, b01, s0);
    s1 = wmma_bf16(a0, b10, s1);
    s1 = wmma_bf16(a1, b11, s1);

    bool v0 = (n0 + lc) < len;
    bool v1 = (n0 + 16 + lc) < len;
    float sc[8];
#pragma unroll
    for (int i = 0; i < 8; i++) {
      float x0 = v0 ? s0[i] : -1e9f;
      float x1 = v1 ? s1[i] : -1e9f;
      float mx = fmaxf(x0, x1);
#pragma unroll
      for (int d = 1; d < 16; d <<= 1) mx = fmaxf(mx, __shfl_xor(mx, d, 16));
      float nm = fmaxf(rm[i], mx);
      sc[i] = __expf(rm[i] - nm);
      rm[i] = nm;
      float p0 = __expf(x0 - nm);
      float p1 = __expf(x1 - nm);
      float s = p0 + p1;
#pragma unroll
      for (int d = 1; d < 16; d <<= 1) s += __shfl_xor(s, d, 16);
      rl[i] = rl[i] * sc[i] + s;
      // stash probs (row = i + hi*8) into per-wave LDS tile (bf16)
      myp[(i + hi * 8) * 40 + lc]      = (__bf16)p0;
      myp[(i + hi * 8) * 40 + 16 + lc] = (__bf16)p1;
    }

#pragma unroll
    for (int j = 0; j < 4; j++)
#pragma unroll
      for (int e = 0; e < 8; e++) o[j][e] *= sc[e];

    bf16x16 pa = wmma_ld_a(myp, 40);   // in-order LDS per wave: no barrier needed
#pragma unroll
    for (int j = 0; j < 4; j++) {
      bf16x16 bv = wmma_ld_b(vh + (size_t)(j * 16) * CT + n0, CT);
      o[j] = wmma_bf16(pa, bv, o[j]);
    }
  }

  // Write attention output (bf16, head-interleaved [B,T,D]); zero invalid rows.
#pragma unroll
  for (int j = 0; j < 4; j++) {
#pragma unroll
    for (int e = 0; e < 8; e++) {
      int row = hi * 8 + e;
      int t = m0 + row;
      float v = (t < len) ? (o[j][e] / rl[e]) : 0.0f;
      aout[((size_t)b * CT + t) * CD + h * CDH + j * 16 + lc] = (__bf16)v;
    }
  }
}

// ---------------------------------------------------------------------------
// Residual + LayerNorm (+mask). FIRST also writes a bf16 copy for next GEMM.
// One block per row.
// ---------------------------------------------------------------------------
template <bool FIRST>
__global__ __launch_bounds__(256) void ln_kernel(
    const float* __restrict__ xa, const float* __restrict__ xb,
    const float* __restrict__ g, const float* __restrict__ be,
    const int* __restrict__ lengths,
    float* __restrict__ outf, __bf16* __restrict__ outb) {
  __shared__ float red0[256];
  __shared__ float red1[256];
  int r = blockIdx.x;
  int b = r >> 11, t = r & (CT - 1);
  float maskv = (t < lengths[b]) ? 1.0f : 0.0f;

  float vloc[3];
  float s = 0.0f, s2 = 0.0f;
#pragma unroll
  for (int it = 0; it < 3; ++it) {
    int i = threadIdx.x + it * 256;
    float v = xa[(size_t)r * CD + i] + xb[(size_t)r * CD + i];
    vloc[it] = v;
    s += v;
    s2 += v * v;
  }
  red0[threadIdx.x] = s;
  red1[threadIdx.x] = s2;
  __syncthreads();
  for (int off = 128; off > 0; off >>= 1) {
    if (threadIdx.x < off) {
      red0[threadIdx.x] += red0[threadIdx.x + off];
      red1[threadIdx.x] += red1[threadIdx.x + off];
    }
    __syncthreads();
  }
  float mean = red0[0] * (1.0f / CD);
  float var = red1[0] * (1.0f / CD) - mean * mean;
  float rstd = rsqrtf(var + 1e-5f);
#pragma unroll
  for (int it = 0; it < 3; ++it) {
    int i = threadIdx.x + it * 256;
    float o = ((vloc[it] - mean) * rstd * g[i] + be[i]) * maskv;
    outf[(size_t)r * CD + i] = o;
    if (FIRST) outb[(size_t)r * CD + i] = (__bf16)o;
  }
}

// ---------------------------------------------------------------------------
extern "C" void kernel_launch(void* const* d_in, const int* in_sizes, int n_in,
                              void* d_out, int out_size, void* d_ws, size_t ws_size,
                              hipStream_t stream) {
  (void)in_sizes; (void)n_in; (void)out_size; (void)ws_size;
  const float* x       = (const float*)d_in[0];
  const int*   lengths = (const int*)d_in[1];
  const float* w_qkv   = (const float*)d_in[2];
  const float* b_qkv   = (const float*)d_in[3];
  const float* w_proj  = (const float*)d_in[4];
  const float* b_proj  = (const float*)d_in[5];
  const float* ln1_g   = (const float*)d_in[6];
  const float* ln1_b   = (const float*)d_in[7];
  const float* w_fc    = (const float*)d_in[8];
  const float* b_fc    = (const float*)d_in[9];
  const float* w_out   = (const float*)d_in[10];
  const float* b_out   = (const float*)d_in[11];
  const float* ln2_g   = (const float*)d_in[12];
  const float* ln2_b   = (const float*)d_in[13];
  float* out = (float*)d_out;

  char* ws = (char*)d_ws;
  size_t off = 0;
  auto salloc = [&](size_t bytes) -> char* {
    char* p = ws + off;
    off += (bytes + 255) & ~(size_t)255;
    return p;
  };
  __bf16* x_bf   = (__bf16*)salloc((size_t)CM * CD * 2);
  __bf16* wtqkv  = (__bf16*)salloc((size_t)3 * CD * CD * 2);
  __bf16* wtproj = (__bf16*)salloc((size_t)CD * CD * 2);
  __bf16* wtfc   = (__bf16*)salloc((size_t)4 * CD * CD * 2);
  __bf16* wtout  = (__bf16*)salloc((size_t)4 * CD * CD * 2);
  __bf16* q_bf   = (__bf16*)salloc((size_t)CB * CH * CT * CDH * 2);
  __bf16* k_bf   = (__bf16*)salloc((size_t)CB * CH * CT * CDH * 2);
  __bf16* v_bf   = (__bf16*)salloc((size_t)CB * CH * CDH * CT * 2);
  __bf16* a_bf   = (__bf16*)salloc((size_t)CM * CD * 2);
  float*  y1     = (float*)salloc((size_t)CM * CD * 4);
  float*  n_f    = (float*)salloc((size_t)CM * CD * 4);
  __bf16* n_bf   = (__bf16*)salloc((size_t)CM * CD * 2);
  __bf16* g_bf   = (__bf16*)salloc((size_t)CM * 4 * CD * 2);
  float*  y2     = (float*)salloc((size_t)CM * CD * 4);

  // 1) Convert activations + weights (weights transposed to [N][K] bf16)
  {
    int n = CM * CD;
    cvt_bf16<<<(n + 255) / 256, 256, 0, stream>>>(x, x_bf, n);
  }
  transpose_cvt<<<(3 * CD * CD + 255) / 256, 256, 0, stream>>>(w_qkv, wtqkv, CD, 3 * CD);
  transpose_cvt<<<(CD * CD + 255) / 256, 256, 0, stream>>>(w_proj, wtproj, CD, CD);
  transpose_cvt<<<(4 * CD * CD + 255) / 256, 256, 0, stream>>>(w_fc, wtfc, CD, 4 * CD);
  transpose_cvt<<<(4 * CD * CD + 255) / 256, 256, 0, stream>>>(w_out, wtout, 4 * CD, CD);

  // 2) QKV GEMM: [4096,768] x [768,2304], split epilogue
  gemm256<0><<<dim3((3 * CD) / 256, CM / 128), 256, 0, stream>>>(
      x_bf, wtqkv, b_qkv, nullptr, nullptr, q_bf, k_bf, v_bf, CM, 3 * CD, CD);

  // 3) Flash attention: 2*12*128 = 3072 waves, 8 waves/block
  attn_kernel<<<(CB * CH * (CT / 16)) / 8, 256, 0, stream>>>(q_bf, k_bf, v_bf,
                                                             lengths, a_bf);

  // 4) Proj GEMM -> y1 (f32)
  gemm256<1><<<dim3(CD / 256, CM / 128), 256, 0, stream>>>(
      a_bf, wtproj, b_proj, y1, nullptr, nullptr, nullptr, nullptr, CM, CD, CD);

  // 5) n = LN(x + y1) * mask  (f32 + bf16 copies)
  ln_kernel<true><<<CM, 256, 0, stream>>>(x, y1, ln1_g, ln1_b, lengths, n_f, n_bf);

  // 6) FC GEMM + GELU -> g_bf (bf16)
  gemm256<2><<<dim3((4 * CD) / 256, CM / 128), 256, 0, stream>>>(
      n_bf, wtfc, b_fc, nullptr, g_bf, nullptr, nullptr, nullptr, CM, 4 * CD, CD);

  // 7) OUT GEMM -> y2 (f32)
  gemm256<3><<<dim3(CD / 256, CM / 128), 256, 0, stream>>>(
      g_bf, wtout, b_out, y2, nullptr, nullptr, nullptr, nullptr, CM, CD, 4 * CD);

  // 8) h = LN(n + y2) * mask -> d_out
  ln_kernel<false><<<CM, 256, 0, stream>>>(n_f, y2, ln2_g, ln2_b, lengths, out,
                                           nullptr);
}
